// SelfAttention_89515708384010
// MI455X (gfx1250) — matrix operand order repaired and checked by reading, not compiled, not run on hardware
//
#include <hip/hip_runtime.h>

// ---------------------------------------------------------------------------
// Dual cross-attention for MI455X (gfx1250), wave32 + WMMA f16 (f32 accum).
// B=4, N=2048, C=256, H=8, D=32. Flash-style fused attention; K/V panels
// double-buffered in LDS via GLOBAL_LOAD_ASYNC_TO_LDS. All GEMM fragments except V (strided gather)
// are contiguous b128 loads (fp16 panels prepared once).
// ---------------------------------------------------------------------------

typedef _Float16 v16h __attribute__((ext_vector_type(16)));
typedef _Float16 v8h  __attribute__((ext_vector_type(8)));
typedef _Float16 v4h  __attribute__((ext_vector_type(4)));
typedef float    v8f  __attribute__((ext_vector_type(8)));
typedef float    v4f  __attribute__((ext_vector_type(4)));
typedef int      v2i  __attribute__((ext_vector_type(2)));

#define BQ 4
#define NQ 2048
#define CQ 256
#define HQ 8
#define DQ 32
#define SCALEQ 0.17677669529663687f  // 1/sqrt(32)

#define AS1 __attribute__((address_space(1)))
#define AS3 __attribute__((address_space(3)))

#if defined(__has_builtin)
#if __has_builtin(__builtin_amdgcn_global_load_async_to_lds_b64) && \
    __has_builtin(__builtin_amdgcn_s_wait_asynccnt)
#define USE_ASYNC 1
#endif
#endif
#ifndef USE_ASYNC
#define USE_ASYNC 0
#endif

// 8-byte copy global -> LDS (async when the CDNA5 builtin exists).
__device__ __forceinline__ void cp8_g2l(_Float16* lds, const _Float16* g) {
#if USE_ASYNC
  __builtin_amdgcn_global_load_async_to_lds_b64((AS1 v2i*)(g), (AS3 v2i*)(lds), 0, 0);
#else
  *(v4h*)lds = *(const v4h*)g;
#endif
}

__device__ __forceinline__ void wait_async() {
#if USE_ASYNC
  __builtin_amdgcn_s_wait_asynccnt(0);
#endif
}

// ---- WMMA fragment loaders (contiguous k = 16*half + s, same in A and B) ------

// A-matrix 16x32 f16: per lane two contiguous 16B runs -> two b128 loads.
// Requires rows 16B-aligned. lane: half = lane>>4, row m = lane&15.
__device__ __forceinline__ v16h load_a_f16(const _Float16* a, int lda, int lane) {
  const int lh = (lane >> 4) & 1, m = lane & 15;
  const _Float16* row = a + (size_t)m * lda + lh * 16;
  union { v16h v; v8h q[2]; } u;
  u.q[0] = *(const v8h*)(row);       // K = lh*16 .. lh*16+7
  u.q[1] = *(const v8h*)(row + 8);   // K = lh*16+8 .. lh*16+15
  return u.v;
}

// B-matrix 32x16 f16 where element (k,n) lives at base[n*ld + k]
// (transposed-panel storage). One contiguous 32B read per lane (2x b128).
__device__ __forceinline__ v16h load_b_rows(const _Float16* base, int ld, int lane) {
  const int lh = lane >> 4, n = lane & 15;
  const _Float16* p = base + (size_t)n * ld + lh * 16;
  union { v16h v; v8h q[2]; } u;
  u.q[0] = *(const v8h*)(p);
  u.q[1] = *(const v8h*)(p + 8);
  return u.v;
}

// B-matrix 32x16 f16 from row-major [k][n] storage (strided column gather).
__device__ __forceinline__ v16h load_b_cols(const _Float16* base, int ld, int lane) {
  const int lh = lane >> 4, n = lane & 15;
  v16h f{};
#pragma unroll
  for (int v = 0; v < 8; ++v) {
    const int k = lh * 16 + 2 * v;
    f[2 * v]     = base[(size_t)k * ld + n];
    f[2 * v + 1] = base[(size_t)(k + 1) * ld + n];
  }
  return f;
}

__device__ __forceinline__ v8f wmma_f16(v16h a, v16h b, v8f c) {
  return __builtin_amdgcn_wmma_f32_16x16x32_f16(false, a, false, b, (short)0, c,
                                                false, false);
}

// ---------------------------------------------------------------------------
// Kernel P0: x,y -> fp16 copy (8 elements/thread, b128 in / b128 out).
// ---------------------------------------------------------------------------
__global__ void cvt_kernel(const float* __restrict__ x, const float* __restrict__ y,
                           _Float16* __restrict__ xh) {
  const size_t t = (size_t)blockIdx.x * 256 + threadIdx.x;  // 524288 threads
  const size_t base = t * 8;
  const size_t half = (size_t)BQ * NQ * CQ;                 // 2097152
  const float* src = (base < half) ? x : y;
  const size_t off = (base < half) ? base : base - half;
  v4f a0 = *(const v4f*)(src + off);
  v4f a1 = *(const v4f*)(src + off + 4);
  v8h o;
#pragma unroll
  for (int i = 0; i < 4; ++i) { o[i] = (_Float16)a0[i]; o[4 + i] = (_Float16)a1[i]; }
  *(v8h*)(xh + base) = o;
}

// ---------------------------------------------------------------------------
// Kernel P1: transpose + fp16-convert the shared weights (once per launch).
// ---------------------------------------------------------------------------
__global__ void wtrans_kernel(const float* __restrict__ wqkv, const float* __restrict__ wproj,
                              _Float16* __restrict__ wqkvT, _Float16* __restrict__ wprojT) {
  const int t = blockIdx.x * 256 + threadIdx.x;  // 262144 total
  if (t < 768 * 256) {
    const int j = t >> 8, i = t & 255;           // wqkvT[j][i] = wqkv[i][j]
    wqkvT[t] = (_Float16)wqkv[(size_t)i * 768 + j];
  } else {
    const int u = t - 768 * 256;
    const int j = u >> 8, i = u & 255;           // wprojT[j][i] = wproj[i][j]
    wprojT[u] = (_Float16)wproj[(size_t)i * 256 + j];
  }
}

// ---------------------------------------------------------------------------
// Kernel 1: fused QKV projection for both sources (fp16 x fp16 -> fp16).
// qkv layout: [src][b][h][{q,k,v}][n][d], q pre-scaled by 1/sqrt(D).
// ---------------------------------------------------------------------------
__global__ void qkv_kernel(const _Float16* __restrict__ xh,
                           const _Float16* __restrict__ wqkvT, _Float16* __restrict__ qkv) {
  const int tid = threadIdx.x, lane = tid & 31, wave = tid >> 5;
  const int gw = blockIdx.x * 8 + wave;     // 49152 tiles total
  const int mtile = gw / 48, ntile = gw % 48;
  const int row0 = mtile * 16;
  const int src = row0 >> 13;               // 0: x, 1: y
  const int lr  = row0 & 8191;
  const _Float16* in = xh + (size_t)src * BQ * NQ * CQ;
  const int n0 = ntile * 16;

  v8f acc = {};
#pragma unroll
  for (int kk = 0; kk < 8; ++kk) {
    const int k0 = kk * 32;
    v16h a = load_a_f16(in + (size_t)lr * CQ + k0, CQ, lane);
    v16h b = load_b_rows(wqkvT + (size_t)n0 * CQ + k0, CQ, lane);
    acc = wmma_f16(a, b, acc);
  }

  const int mat = n0 >> 8;                  // 0=q 1=k 2=v
  const int cl  = n0 & 255;
  const int hd  = cl >> 5;
  const int d0  = cl & 31;                  // 0 or 16
  const float scale = (mat == 0) ? SCALEQ : 1.0f;
  const int bidx = lr >> 11, nrow0 = lr & 2047;

  _Float16* dst = qkv + ((((size_t)(src * BQ + bidx) * HQ + hd) * 3 + mat) * NQ) * DQ;
  const int lh = lane >> 4, n = lane & 15;
#pragma unroll
  for (int r = 0; r < 8; ++r) {
    const int nr = nrow0 + r + 8 * lh;
    dst[(size_t)nr * DQ + d0 + n] = (_Float16)(acc[r] * scale);
  }
}

// ---------------------------------------------------------------------------
// Kernel 2: flash attention. Branch 0: Q,K from x ; V from y (cross).
//           Branch 1: Q,K from y ; V from x.
// K/V 32-row panels double-buffered in LDS, filled by async copies that
// overlap the WMMA work on the current panel.
// ---------------------------------------------------------------------------
#define KLD 40   // K panel stride (halves): 16B aligned rows, conflict-free b128
#define VLD 36   // V panel stride (halves): 8B aligned rows, disjoint bank halves
#define PLD 40   // P staging stride (halves): 16B aligned rows for b128 A-reads

__global__ void attn_kernel(const _Float16* __restrict__ qkv, _Float16* __restrict__ O) {
  __shared__ alignas(16) _Float16 Kl[2][32][KLD];
  __shared__ alignas(16) _Float16 Vl[2][32][VLD];
  __shared__ alignas(16) _Float16 Pw[8][16][PLD];

  const int tid = threadIdx.x, lane = tid & 31, wave = tid >> 5;
  const int lh = lane >> 4, n = lane & 15;

  const int branch = blockIdx.x / (BQ * HQ);
  const int bh = blockIdx.x % (BQ * HQ);
  const int b = bh / HQ, hd = bh % HQ;
  const int srcqk = branch, srcv = 1 - branch;

  const _Float16* qp = qkv + ((((size_t)(srcqk * BQ + b) * HQ + hd) * 3 + 0) * NQ) * DQ;
  const _Float16* kp = qkv + ((((size_t)(srcqk * BQ + b) * HQ + hd) * 3 + 1) * NQ) * DQ;
  const _Float16* vp = qkv + ((((size_t)(srcv  * BQ + b) * HQ + hd) * 3 + 2) * NQ) * DQ;
  _Float16* Od = O + (((size_t)(branch * BQ + b) * HQ + hd) * NQ) * DQ;

  const int m0 = blockIdx.y * 128 + wave * 16;
  const v16h qf = load_a_f16(qp + (size_t)m0 * DQ, DQ, lane);

  v8f acc0 = {}, acc1 = {};
  float mrow[8], lrow[8];
#pragma unroll
  for (int r = 0; r < 8; ++r) { mrow[r] = -3.0e38f; lrow[r] = 0.0f; }

  // panel fill: 256 threads x 8B for K and V each (32 rows x 64B)
  const int frow = tid >> 3, fseg = (tid & 7) * 4;

  // preload block 0
  cp8_g2l(&Kl[0][frow][fseg], kp + (size_t)frow * DQ + fseg);
  cp8_g2l(&Vl[0][frow][fseg], vp + (size_t)frow * DQ + fseg);
  wait_async();
  __syncthreads();

  const int NKB = NQ / 32;  // 64
  for (int kb = 0; kb < NKB; ++kb) {
    const int cur = kb & 1;
    if (kb + 1 < NKB) {  // async-fill next buffer while computing this one
      const size_t goff = (size_t)((kb + 1) * 32 + frow) * DQ + fseg;
      cp8_g2l(&Kl[cur ^ 1][frow][fseg], kp + goff);
      cp8_g2l(&Vl[cur ^ 1][frow][fseg], vp + goff);
    }

    // S = Q * K^T : contiguous 32B b128 fragment reads from LDS K panel
    v16h bk0 = load_b_rows(&Kl[cur][0][0], KLD, lane);
    v16h bk1 = load_b_rows(&Kl[cur][16][0], KLD, lane);
    v8f z = {};
    v8f s0 = wmma_f16(qf, bk0, z);
    v8f s1 = wmma_f16(qf, bk1, z);

    // online softmax over this 32-wide block (row = r + 8*lh, cols across lanes)
#pragma unroll
    for (int r = 0; r < 8; ++r) {
      float sm = fmaxf(s0[r], s1[r]);
      sm = fmaxf(sm, __shfl_xor(sm, 1, 32));
      sm = fmaxf(sm, __shfl_xor(sm, 2, 32));
      sm = fmaxf(sm, __shfl_xor(sm, 4, 32));
      sm = fmaxf(sm, __shfl_xor(sm, 8, 32));
      const float mn = fmaxf(mrow[r], sm);
      const float al = __expf(mrow[r] - mn);
      const float p0 = __expf(s0[r] - mn);
      const float p1 = __expf(s1[r] - mn);
      float rs = p0 + p1;
      rs += __shfl_xor(rs, 1, 32);
      rs += __shfl_xor(rs, 2, 32);
      rs += __shfl_xor(rs, 4, 32);
      rs += __shfl_xor(rs, 8, 32);
      lrow[r] = lrow[r] * al + rs;
      mrow[r] = mn;
      acc0[r] *= al;
      acc1[r] *= al;
      Pw[wave][r + 8 * lh][n]      = (_Float16)p0;
      Pw[wave][r + 8 * lh][16 + n] = (_Float16)p1;
    }

    // P (C-layout) -> A-layout via per-wave LDS, then O += P * V
    v16h pa  = load_a_f16(&Pw[wave][0][0], PLD, lane);
    v16h bv0 = load_b_cols(&Vl[cur][0][0], VLD, lane);       // d = n
    v16h bv1 = load_b_cols(&Vl[cur][0][0] + 16, VLD, lane);  // d = 16 + n
    acc0 = wmma_f16(pa, bv0, acc0);
    acc1 = wmma_f16(pa, bv1, acc1);

    wait_async();     // own async fills done before the barrier
    __syncthreads();  // everyone done filling next / reading cur
  }

  // normalize and store O (fp16, [n][d])
#pragma unroll
  for (int r = 0; r < 8; ++r) {
    const float inv = 1.0f / lrow[r];
    const int row = m0 + r + 8 * lh;
    Od[(size_t)row * DQ + n]      = (_Float16)(acc0[r] * inv);
    Od[(size_t)row * DQ + 16 + n] = (_Float16)(acc1[r] * inv);
  }
}

// ---------------------------------------------------------------------------
// Kernel 3: out = O @ w_proj + b_proj + residual, fp32 output.
// O A-fragments: each k-subtile (32 cols) lies inside one head's contiguous
// 32-half row, so load_a_f16 works directly per head block.
// ---------------------------------------------------------------------------
__global__ void proj_kernel(const float* __restrict__ x, const float* __restrict__ y,
                            const _Float16* __restrict__ wprojT, const float* __restrict__ bproj,
                            const _Float16* __restrict__ O, float* __restrict__ out) {
  const int tid = threadIdx.x, lane = tid & 31, wave = tid >> 5;
  const int gw = blockIdx.x * 8 + wave;     // 16384 tiles
  const int mtile = gw >> 4, ntile = gw & 15;
  const int row0 = mtile * 16;
  const int src = row0 >> 13;
  const int lr  = row0 & 8191;
  const int b = lr >> 11, nr0 = lr & 2047;

  const _Float16* Ob = O + ((size_t)(src * BQ + b) * HQ) * NQ * DQ;

  v8f acc = {};
#pragma unroll
  for (int kk = 0; kk < 8; ++kk) {          // head kk supplies k-cols kk*32..+31
    v16h a  = load_a_f16(Ob + ((size_t)kk * NQ + nr0) * DQ, DQ, lane);
    v16h bb = load_b_rows(wprojT + (size_t)(ntile * 16) * CQ + kk * 32, CQ, lane);
    acc = wmma_f16(a, bb, acc);
  }

  const float* res = (src ? y : x) + ((size_t)b * NQ + nr0) * CQ;
  float* op = out + (size_t)src * BQ * NQ * CQ + ((size_t)b * NQ + nr0) * CQ;
  const int lh = lane >> 4, n = lane & 15;
  const int c = ntile * 16 + n;
  const float bias = bproj[c];
#pragma unroll
  for (int r = 0; r < 8; ++r) {
    const int m = r + 8 * lh;
    op[(size_t)m * CQ + c] = acc[r] + bias + res[(size_t)m * CQ + c];
  }
}

// ---------------------------------------------------------------------------
extern "C" void kernel_launch(void* const* d_in, const int* in_sizes, int n_in,
                              void* d_out, int out_size, void* d_ws, size_t ws_size,
                              hipStream_t stream) {
  const float* x     = (const float*)d_in[0];
  const float* y     = (const float*)d_in[1];
  const float* wqkv  = (const float*)d_in[2];
  const float* wproj = (const float*)d_in[3];
  const float* bproj = (const float*)d_in[4];
  float* out = (float*)d_out;

  _Float16* qkv    = (_Float16*)d_ws;                           // 24 MiB
  _Float16* O      = qkv + (size_t)2 * BQ * HQ * 3 * NQ * DQ;   // 8 MiB
  _Float16* wqkvT  = O + (size_t)2 * BQ * HQ * NQ * DQ;         // 384 KiB
  _Float16* wprojT = wqkvT + (size_t)768 * 256;                 // 128 KiB
  _Float16* xh     = wprojT + (size_t)256 * 256;                // 8 MiB (fp16 x|y)

  cvt_kernel   <<<dim3(2048),                 256, 0, stream>>>(x, y, xh);
  wtrans_kernel<<<dim3(1024),                 256, 0, stream>>>(wqkv, wproj, wqkvT, wprojT);
  qkv_kernel   <<<dim3(6144),                 256, 0, stream>>>(xh, wqkvT, qkv);
  attn_kernel  <<<dim3(2 * BQ * HQ, NQ / 128), 256, 0, stream>>>(qkv, O);
  proj_kernel  <<<dim3(2048),                 256, 0, stream>>>(x, y, wprojT, bproj, O, out);
}
